// RNNet_41575283425795
// MI455X (gfx1250) — matrix operand order repaired and checked by reading, not compiled, not run on hardware
//
#include <hip/hip_runtime.h>
#include <hip/hip_bf16.h>

// ---------------------------------------------------------------------------
// CDNA5 (gfx1250) persistent GRU forecaster.
//  B=256, CTX=512, PRED=64, H=512.  16 WGs x 512 threads (16 waves, wave32).
//  Each WG owns 16 batch rows; wave w owns gate tiles == w (mod 16).
//  GEMMs: v_wmma_f32_16x16x32_bf16, M=16 rows, N=1536 gates, K=512.
//  Register-pressure-safe software pipelining:
//   - column tiles (i=0,1) processed sequentially, gates combined per half
//   - rolled kt loops with unroll == ring depth (4 / 2) + peeled epilogue:
//     bounded scheduler window -> no spills, 4-triple prefetch distance
//   - weight fragment addresses: one base pointer + immediate offsets
// ---------------------------------------------------------------------------

typedef __attribute__((ext_vector_type(16))) __bf16 v16bf;
typedef __attribute__((ext_vector_type(8)))  __bf16 v8bf;
typedef __attribute__((ext_vector_type(8)))  float  v8f;

#define HID    512
#define HPAD   520                  // padded LDS row stride (bf16 elements)
#define KT     16                   // K tiles of 32
#define NT     96                   // N tiles of 16 (1536/16)
#define CTXL   512
#define PREDL  64
#define NTSTR  (KT * 512)           // 8192 elements per n-tile in packed W
#define GSTR   (32 * NTSTR)         // 262144: +1 gate group (32 n-tiles)
#define WELEM  (NT * NTSTR)         // 786432 packed bf16 per weight matrix

// ---- math helpers ---------------------------------------------------------
__device__ __forceinline__ float sigmoidf_(float x) {
  return 1.0f / (1.0f + __expf(-x));
}
__device__ __forceinline__ float tanhf_(float x) {
  x = fminf(fmaxf(x, -15.0f), 15.0f);
  float e = __expf(2.0f * x);
  return (e - 1.0f) / (e + 1.0f);
}

// ---- WMMA A fragment (16x32 bf16) from padded LDS h buffer ----------------
__device__ __forceinline__ v16bf loadA(const __bf16* hb, int kt, int lane) {
  int base = (lane & 15) * HPAD + kt * 32 + ((lane >> 4) << 3);
  const v8bf* p0 = (const v8bf*)(hb + base);
  const v8bf* p1 = (const v8bf*)(hb + base + 16);
  v8bf lo = *p0, hi = *p1;
  return __builtin_shufflevector(lo, hi, 0,1,2,3,4,5,6,7,8,9,10,11,12,13,14,15);
}

__device__ __forceinline__ v16bf ldfrag(const __bf16* base, int off) {
  return *(const v16bf*)(base + off);
}

__device__ __forceinline__ v8f wmma_bf16(v16bf a, v16bf b, v8f c) {
  return __builtin_amdgcn_wmma_f32_16x16x32_bf16(
      false, a, false, b, (short)0, c, false, false);
}

// ---- layer-0 half GEMM: one column tile, 3 accumulators, ring depth 4 -----
__device__ __forceinline__ void gemm_l0_half(
    v8f* acc /*r,z,n*/, const __bf16* h0b, const __bf16* wb, int lane) {
  v16bf B[4][3];
#pragma unroll
  for (int kt = 0; kt < 4; ++kt)
#pragma unroll
    for (int g = 0; g < 3; ++g) B[kt][g] = ldfrag(wb, g * GSTR + kt * 512);
#pragma unroll 4
  for (int kt = 0; kt < KT - 4; ++kt) {   // rolled x4: bounded live ranges
    const int q = kt & 3;
    v16bf a = loadA(h0b, kt, lane);
    acc[0] = wmma_bf16(a, B[q][0], acc[0]);
    acc[1] = wmma_bf16(a, B[q][1], acc[1]);
    acc[2] = wmma_bf16(a, B[q][2], acc[2]);
#pragma unroll
    for (int g = 0; g < 3; ++g) B[q][g] = ldfrag(wb, g * GSTR + (kt + 4) * 512);
  }
#pragma unroll
  for (int kt = KT - 4; kt < KT; ++kt) {  // epilogue: no reload
    const int q = kt & 3;
    v16bf a = loadA(h0b, kt, lane);
    acc[0] = wmma_bf16(a, B[q][0], acc[0]);
    acc[1] = wmma_bf16(a, B[q][1], acc[1]);
    acc[2] = wmma_bf16(a, B[q][2], acc[2]);
  }
}

// ---- layer-1 half GEMM: one column tile, 4 accumulators, ring depth 2x2 ---
__device__ __forceinline__ void gemm_l1_half(
    v8f* acc /*jr,jz,nx,nh*/, const __bf16* h0b, const __bf16* h1b,
    const __bf16* bx, const __bf16* bh, int lane) {
  v16bf B[4][3];                          // [ (kt&1)*2 + src ][gate]
#pragma unroll
  for (int kt = 0; kt < 2; ++kt)
#pragma unroll
    for (int g = 0; g < 3; ++g) {
      B[kt * 2 + 0][g] = ldfrag(bx, g * GSTR + kt * 512);
      B[kt * 2 + 1][g] = ldfrag(bh, g * GSTR + kt * 512);
    }
#pragma unroll 2
  for (int kt = 0; kt < KT - 2; ++kt) {   // rolled x2
    const int qb = (kt & 1) * 2;
    v16bf a0 = loadA(h0b, kt, lane);
    acc[0] = wmma_bf16(a0, B[qb][0], acc[0]);
    acc[1] = wmma_bf16(a0, B[qb][1], acc[1]);
    acc[2] = wmma_bf16(a0, B[qb][2], acc[2]);
#pragma unroll
    for (int g = 0; g < 3; ++g) B[qb][g] = ldfrag(bx, g * GSTR + (kt + 2) * 512);
    v16bf a1 = loadA(h1b, kt, lane);
    acc[0] = wmma_bf16(a1, B[qb + 1][0], acc[0]);
    acc[1] = wmma_bf16(a1, B[qb + 1][1], acc[1]);
    acc[3] = wmma_bf16(a1, B[qb + 1][2], acc[3]);
#pragma unroll
    for (int g = 0; g < 3; ++g) B[qb + 1][g] = ldfrag(bh, g * GSTR + (kt + 2) * 512);
  }
#pragma unroll
  for (int kt = KT - 2; kt < KT; ++kt) {  // epilogue
    const int qb = (kt & 1) * 2;
    v16bf a0 = loadA(h0b, kt, lane);
    acc[0] = wmma_bf16(a0, B[qb][0], acc[0]);
    acc[1] = wmma_bf16(a0, B[qb][1], acc[1]);
    acc[2] = wmma_bf16(a0, B[qb][2], acc[2]);
    v16bf a1 = loadA(h1b, kt, lane);
    acc[0] = wmma_bf16(a1, B[qb + 1][0], acc[0]);
    acc[1] = wmma_bf16(a1, B[qb + 1][1], acc[1]);
    acc[3] = wmma_bf16(a1, B[qb + 1][2], acc[3]);
  }
}

// ---- GRU layer 0 step (scalar input per row; rank-1 xg folded) ------------
// c[i*7+]: 0=wr 1=wz 2=wn 3=br(bi+bh) 4=bz(bi+bh) 5=bin 6=bhn
__device__ __forceinline__ void gru_layer0(
    int w, int lane, float* hp0, __bf16* h0b, const float* xbuf,
    const __bf16* Whh0p, const float* c /*[14]*/) {
  const __bf16* wb = Whh0p + lane * 16 + w * NTSTR;
  const int hs = lane >> 4, nl = lane & 15;
  float hn[16];
#pragma unroll
  for (int i = 0; i < 2; ++i) {           // sequential column tiles
    v8f acc[3] = {(v8f){}, (v8f){}, (v8f){}};
    gemm_l0_half(acc, h0b, wb + i * (16 * NTSTR), lane);
    const float* ci = c + i * 7;
#pragma unroll
    for (int rr = 0; rr < 8; ++rr) {
      float xv = xbuf[hs * 8 + rr];
      float r = sigmoidf_(xv * ci[0] + acc[0][rr] + ci[3]);
      float z = sigmoidf_(xv * ci[1] + acc[1][rr] + ci[4]);
      float n = tanhf_(xv * ci[2] + ci[5] + r * (acc[2][rr] + ci[6]));
      float v = (1.0f - z) * n + z * hp0[i * 8 + rr];
      hp0[i * 8 + rr] = v;
      hn[i * 8 + rr] = v;
    }
  }
  __syncthreads();                        // all reads of h0b done
#pragma unroll
  for (int i = 0; i < 2; ++i) {
    int col = (w + 16 * i) * 16 + nl;
#pragma unroll
    for (int rr = 0; rr < 8; ++rr)
      h0b[(hs * 8 + rr) * HPAD + col] = (__bf16)hn[i * 8 + rr];
  }
  __syncthreads();                        // new h0 visible
}

// ---- GRU layer 1 step -----------------------------------------------------
// cb[i*4+]: 0=br 1=bz 2=bixn 3=bhn
__device__ __forceinline__ void gru_layer1(
    int tid, int w, int lane,
    float* hp1, const __bf16* h0b, __bf16* h1b, float* h1f,
    const __bf16* Wih1p, const __bf16* Whh1p, const float* cb /*[8]*/,
    bool fillx, float xnext, float* xbuf) {
  const __bf16* bx = Wih1p + lane * 16 + w * NTSTR;
  const __bf16* bh = Whh1p + lane * 16 + w * NTSTR;
  const int hs = lane >> 4, nl = lane & 15;
  float hn[16];
#pragma unroll
  for (int i = 0; i < 2; ++i) {           // sequential column tiles
    v8f acc[4] = {(v8f){}, (v8f){}, (v8f){}, (v8f){}};
    gemm_l1_half(acc, h0b, h1b, bx + i * (16 * NTSTR), bh + i * (16 * NTSTR), lane);
    const float* ci = cb + i * 4;
#pragma unroll
    for (int rr = 0; rr < 8; ++rr) {
      float r = sigmoidf_(acc[0][rr] + ci[0]);
      float z = sigmoidf_(acc[1][rr] + ci[1]);
      float n = tanhf_(acc[2][rr] + ci[2] + r * (acc[3][rr] + ci[3]));
      float v = (1.0f - z) * n + z * hp1[i * 8 + rr];
      hp1[i * 8 + rr] = v;
      hn[i * 8 + rr] = v;
    }
  }
  __syncthreads();
#pragma unroll
  for (int i = 0; i < 2; ++i) {
    int col = (w + 16 * i) * 16 + nl;
#pragma unroll
    for (int rr = 0; rr < 8; ++rr) {
      int row = hs * 8 + rr;
      h1b[row * HPAD + col] = (__bf16)hn[i * 8 + rr];
      h1f[row * HID + col] = hn[i * 8 + rr];   // f32 copy for output head
    }
  }
  if (fillx && tid < 16) xbuf[tid] = xnext;    // next step's input
  __syncthreads();
}

// ---- head projection: y = h1.Wout + bout; writes output and next x --------
__device__ __forceinline__ void project_head(
    int w, int lane, const float* h1f,
    const float* Wout, const float* bout,
    const float* scalebuf, float* xbuf,
    float* out, int b0, int p) {
  float acc = 0.0f;
  for (int c = lane; c < HID; c += 32)
    acc += h1f[w * HID + c] * Wout[c];
#pragma unroll
  for (int off = 16; off >= 1; off >>= 1)
    acc += __shfl_xor(acc, off, 32);
  if (lane == 0) {
    float y = acc + bout[0];           // y/scale = next GRU input (scale cancels)
    xbuf[w] = y;
    out[(size_t)(b0 + w) * PREDL + p] = y * scalebuf[w];
  }
  __syncthreads();
}

// ---- weight pre-pack: f32 [1536x512] -> bf16 WMMA-B lane layout -----------
__global__ void pack_w(const float* __restrict__ src, __bf16* __restrict__ dst) {
  int idx = blockIdx.x * blockDim.x + threadIdx.x;      // [0, WELEM)
  int j    = idx & 15;
  int lane = (idx >> 4) & 31;
  int kt   = (idx >> 9) & 15;
  int nt   = idx >> 13;
  int n = nt * 16 + (lane & 15);
  int k = kt * 32 + ((lane >> 4) << 3) + (j & 7) + ((j >> 3) << 4);
  dst[idx] = (__bf16)src[n * HID + k];
}

// ---- persistent GRU kernel ------------------------------------------------
__global__ __launch_bounds__(512) void gru_persist(
    const float* __restrict__ ctx,
    const float* __restrict__ Wih0, const float* __restrict__ bih0,
    const float* __restrict__ bhh0,
    const float* __restrict__ bih1, const float* __restrict__ bhh1,
    const float* __restrict__ Wout, const float* __restrict__ bout,
    const __bf16* __restrict__ Whh0p, const __bf16* __restrict__ Wih1p,
    const __bf16* __restrict__ Whh1p,
    float* __restrict__ out) {
  __shared__ __bf16 h0b[16 * HPAD];       // padded bf16 h, WMMA A feed
  __shared__ __bf16 h1b[16 * HPAD];
  __shared__ float  h1f[16 * HID];        // f32 h1 for the output head
  __shared__ float  xbuf[16];
  __shared__ float  scalebuf[16];
  __shared__ float  invs[16];
  __shared__ float  red[16 * 32];

  const int tid  = threadIdx.x;
  const int lane = tid & 31;
  const int w    = tid >> 5;              // wave id 0..15
  const int nl   = lane & 15;
  const int b0   = blockIdx.x * 16;

  float hp0[16], hp1[16];                 // register h_prev (static ownership)
#pragma unroll
  for (int i = 0; i < 16; ++i) { hp0[i] = 0.0f; hp1[i] = 0.0f; }

  // hoisted loop-invariant per-lane scalars (r/z biases pre-summed)
  float l0c[14], l1c[8];
#pragma unroll
  for (int i = 0; i < 2; ++i) {
    int col = (w + 16 * i) * 16 + nl;
    l0c[i*7+0] = Wih0[col]; l0c[i*7+1] = Wih0[col+512]; l0c[i*7+2] = Wih0[col+1024];
    l0c[i*7+3] = bih0[col]       + bhh0[col];
    l0c[i*7+4] = bih0[col+512]   + bhh0[col+512];
    l0c[i*7+5] = bih0[col+1024];
    l0c[i*7+6] = bhh0[col+1024];
    l1c[i*4+0] = bih1[col]       + bhh1[col];
    l1c[i*4+1] = bih1[col+512]   + bhh1[col+512];
    l1c[i*4+2] = bih1[col+1024];
    l1c[i*4+3] = bhh1[col+1024];
  }

  // zero hidden-state LDS
  for (int i = tid; i < 16 * HPAD; i += 512) {
    h0b[i] = (__bf16)0.0f; h1b[i] = (__bf16)0.0f;
  }
  for (int i = tid; i < 16 * HID; i += 512) h1f[i] = 0.0f;

  // scale = max(|mean(context_row)|, 1e-5)
  {
    int row = tid >> 5, s = tid & 31;
    float acc = 0.0f;
    const float* cr = ctx + (size_t)(b0 + row) * CTXL;
    for (int t = s; t < CTXL; t += 32) acc += cr[t];
    red[row * 32 + s] = acc;
  }
  __syncthreads();
  if (tid < 16) {
    float acc = 0.0f;
    for (int s = 0; s < 32; ++s) acc += red[tid * 32 + s];
    float sc = fmaxf(fabsf(acc / (float)CTXL), 1e-5f);
    scalebuf[tid] = sc; invs[tid] = 1.0f / sc;
  }
  __syncthreads();
  if (tid < 16) xbuf[tid] = ctx[(size_t)(b0 + tid) * CTXL] * invs[tid];
  __syncthreads();

  // ---------------- encoder: 512 steps ----------------
  for (int t = 0; t < CTXL; ++t) {
    bool fill = (t + 1 < CTXL);
    float xnext = 0.0f;
    if (fill && tid < 16)
      xnext = ctx[(size_t)(b0 + tid) * CTXL + t + 1] * invs[tid];
    gru_layer0(w, lane, hp0, h0b, xbuf, Whh0p, l0c);
    gru_layer1(tid, w, lane, hp1, h0b, h1b, h1f, Wih1p, Whh1p, l1c,
               fill, xnext, xbuf);
  }

  // ---------------- first forecast + decoder ----------------
  project_head(w, lane, h1f, Wout, bout, scalebuf, xbuf, out, b0, 0);
  for (int p = 1; p < PREDL; ++p) {
    gru_layer0(w, lane, hp0, h0b, xbuf, Whh0p, l0c);
    gru_layer1(tid, w, lane, hp1, h0b, h1b, h1f, Wih1p, Whh1p, l1c,
               false, 0.0f, xbuf);
    project_head(w, lane, h1f, Wout, bout, scalebuf, xbuf, out, b0, p);
  }
}

// ---------------------------------------------------------------------------
extern "C" void kernel_launch(void* const* d_in, const int* in_sizes, int n_in,
                              void* d_out, int out_size, void* d_ws, size_t ws_size,
                              hipStream_t stream) {
  const float* ctx  = (const float*)d_in[0];
  const float* Wih0 = (const float*)d_in[1];
  const float* Whh0 = (const float*)d_in[2];
  const float* bih0 = (const float*)d_in[3];
  const float* bhh0 = (const float*)d_in[4];
  const float* Wih1 = (const float*)d_in[5];
  const float* Whh1 = (const float*)d_in[6];
  const float* bih1 = (const float*)d_in[7];
  const float* bhh1 = (const float*)d_in[8];
  const float* Wout = (const float*)d_in[9];
  const float* bout = (const float*)d_in[10];

  __bf16* ws    = (__bf16*)d_ws;            // 3 x 1.5 MB packed weights
  __bf16* Whh0p = ws;
  __bf16* Wih1p = ws + WELEM;
  __bf16* Whh1p = ws + 2 * (size_t)WELEM;

  pack_w<<<WELEM / 256, 256, 0, stream>>>(Whh0, Whh0p);
  pack_w<<<WELEM / 256, 256, 0, stream>>>(Wih1, Wih1p);
  pack_w<<<WELEM / 256, 256, 0, stream>>>(Whh1, Whh1p);

  gru_persist<<<16, 512, 0, stream>>>(ctx, Wih0, bih0, bhh0, bih1, bhh1,
                                      Wout, bout, Whh0p, Wih1p, Whh1p,
                                      (float*)d_out);
}
